// CVRP_Encoder_56487409877522
// MI455X (gfx1250) — matrix operand order repaired and checked by reading, not compiled, not run on hardware
//
#include <hip/hip_runtime.h>
#include <hip/hip_bf16.h>
#include <hip/hip_fp16.h>

// ---------------------------------------------------------------------------
// CVRP encoder for gfx1250 (MI455X). wave32, WMMA f32_16x16x32_f16.
// Fragments fed by ds_load_b128 pairs (A row-major, B transposed in LDS).
// Pure-copy stagings use GLOBAL_LOAD_ASYNC_TO_LDS_B128 when available.
// ---------------------------------------------------------------------------

typedef __attribute__((ext_vector_type(16))) _Float16 v16h;
typedef __attribute__((ext_vector_type(8)))  _Float16 v8h;
typedef __attribute__((ext_vector_type(4)))  _Float16 v4h;
typedef __attribute__((ext_vector_type(8)))  float    v8f;
typedef __attribute__((ext_vector_type(4)))  int      v4i;
typedef _Float16 h16;

#define NB   32      // batch
#define S    1000    // sequence (N+1)
#define SP   1008    // padded rows (63 * 16)
#define SK   1024    // padded attention K dim (32 * 32)
#define DH   128     // model dim
#define FH   512     // ffn dim
#define NL   6       // layers
#define EPSN 1e-5f
#define LDT  40      // transposed-tile row stride in halves (80 B, 16B-aligned)

#define WMMA_F16(a, b, c) \
  __builtin_amdgcn_wmma_f32_16x16x32_f16(false, (a), false, (b), (short)0, (c), false, false)

// -------- async global->LDS (CDNA5): guarded, with vector-copy fallback ----
#if defined(__has_builtin)
#  if __has_builtin(__builtin_amdgcn_global_load_async_to_lds_b128)
#    define USE_ASYNC_LDS 1
#  endif
#endif
#ifndef USE_ASYNC_LDS
#  define USE_ASYNC_LDS 0
#endif

#if USE_ASYNC_LDS
typedef __attribute__((address_space(1))) v4i gv4i;   // global int4
typedef __attribute__((address_space(3))) v4i lv4i;   // LDS int4
#endif

__device__ __forceinline__ void copy16_g2l(const h16* g, h16* l) {
#if USE_ASYNC_LDS
  __builtin_amdgcn_global_load_async_to_lds_b128((gv4i*)g, (lv4i*)l, 0, 0);
#else
  *reinterpret_cast<uint4*>(l) = *reinterpret_cast<const uint4*>(g);
#endif
}

__device__ __forceinline__ void wait_async0() {
#if USE_ASYNC_LDS
#  if __has_builtin(__builtin_amdgcn_s_wait_asynccnt)
  __builtin_amdgcn_s_wait_asynccnt(0);
#  else
  asm volatile("s_wait_asynccnt 0" ::: "memory");
#  endif
#endif
}

// A-fragment (16x32 f16) from row-major LDS tile: lane m = lane&15; lanes 0-15
// hold K 0-7/16-23, lanes 16-31 hold K 8-15/24-31 -> two contiguous 16B chunks.
__device__ __forceinline__ v16h load_a_frag(const h16* A, int lda, int kc) {
  const int lane = threadIdx.x & 31;
  const int m = lane & 15;
  const int kb = (lane >> 4) << 3;   // 0 or 8
  const h16* p = A + m * lda + kc + kb;
  const v8h lo = *reinterpret_cast<const v8h*>(p);
  const v8h hi = *reinterpret_cast<const v8h*>(p + 16);
  return __builtin_shufflevector(lo, hi, 0, 1, 2, 3, 4, 5, 6, 7,
                                 8, 9, 10, 11, 12, 13, 14, 15);
}

// B-fragment (32x16 f16) from TRANSPOSED LDS tile bT[n][k]: lane n = n0+(lane&15);
// lanes 0-15 hold K 0-7/16-23, lanes 16-31 hold K 8-15/24-31 -> two contiguous 16B chunks.
__device__ __forceinline__ v16h load_bT_frag(const h16* BT, int ldb, int n0) {
  const int lane = threadIdx.x & 31;
  const int n = n0 + (lane & 15);
  const int kb = (lane >> 4) << 3;   // 0 or 8
  const h16* p = BT + n * ldb + kb;
  const v8h lo = *reinterpret_cast<const v8h*>(p);
  const v8h hi = *reinterpret_cast<const v8h*>(p + 16);
  return __builtin_shufflevector(lo, hi, 0, 1, 2, 3, 4, 5, 6, 7,
                                 8, 9, 10, 11, 12, 13, 14, 15);
}

// ---------------------------------------------------------------------------
// Embedding.
// ---------------------------------------------------------------------------
__global__ __launch_bounds__(128) void embed_kernel(
    const float* __restrict__ depot, const float* __restrict__ node,
    const int* __restrict__ flag,
    const float* __restrict__ Wd, const float* __restrict__ bd,
    const float* __restrict__ Wn, const float* __restrict__ bn,
    const float* __restrict__ Win, const float* __restrict__ binv,
    const float* __restrict__ Wout, const float* __restrict__ bout,
    float* __restrict__ X)
{
  const int b = blockIdx.x;
  const int d = threadIdx.x;
  __shared__ float row[DH];
  float* Xb = X + (size_t)b * SP * DH;

  const float dx = depot[b * 2 + 0], dy = depot[b * 2 + 1];
  Xb[d] = Wd[d * 2 + 0] * dx + Wd[d * 2 + 1] * dy + bd[d];

  const float* nb = node + (size_t)b * (S - 1) * 3;
  const float w0 = Wn[d * 3 + 0], w1 = Wn[d * 3 + 1], w2 = Wn[d * 3 + 2], bb = bn[d];
  for (int i = 0; i < S - 1; ++i) {
    Xb[(i + 1) * DH + d] = w0 * nb[i * 3 + 0] + w1 * nb[i * 3 + 1] + w2 * nb[i * 3 + 2] + bb;
  }
  for (int r = S; r < SP; ++r) Xb[r * DH + d] = 0.f;
  __syncthreads();

  // Win on row 1 (node 0)
  row[d] = Xb[DH + d];
  __syncthreads();
  float acc = binv[d];
  for (int k = 0; k < DH; ++k) acc += Win[d * DH + k] * row[k];
  __syncthreads();
  Xb[DH + d] = acc;
  __syncthreads();

  // Wout on row idx (0 if flag==1 else 999; never 1)
  const int idx = (flag[b] == 1) ? 0 : (S - 1);
  row[d] = Xb[idx * DH + d];
  __syncthreads();
  acc = bout[d];
  for (int k = 0; k < DH; ++k) acc += Wout[d * DH + k] * row[k];
  __syncthreads();
  Xb[idx * DH + d] = acc;
}

// Zero EKVt t-range [S, SK) for all 256 channels (feature-major layout).
__global__ __launch_bounds__(256) void ekv_tail_zero_kernel(h16* __restrict__ EKVt) {
  const int b = blockIdx.x;
  h16* p = EKVt + ((size_t)b * 256 + threadIdx.x) * SK;
  const uint4 z = {0u, 0u, 0u, 0u};
  *reinterpret_cast<uint4*>(p + 1000) = z;
  *reinterpret_cast<uint4*>(p + 1008) = z;
  *reinterpret_cast<uint4*>(p + 1016) = z;
}

// ---------------------------------------------------------------------------
// QKV: SIGQ[b][s][d] = sigmoid(x@Wq.T) f16 ;
//      EKVt[b][c][t] feature-major: c<128 -> exp(k)*v, c>=128 -> exp(k).
// Block: 16 rows x 128 cols, 8 waves x 1 N-tile, K = 128 in 4 chunks.
// ---------------------------------------------------------------------------
__global__ __launch_bounds__(256) void qkv_kernel(
    const float* __restrict__ X,
    const float* __restrict__ Wq, const float* __restrict__ Wk,
    const float* __restrict__ Wv,
    h16* __restrict__ SIGQ, h16* __restrict__ EKVt)
{
  const int tile = blockIdx.x, b = blockIdx.y;
  const int s0 = tile * 16;
  const int tid = threadIdx.x;
  const int wave = tid >> 5, lane = tid & 31;

  __shared__ h16 xa[16][DH];
  __shared__ h16 wbT[3][DH][LDT];   // transposed: [n][k]

  const float* Xb = X + ((size_t)b * SP + s0) * DH;
  for (int i = tid; i < 16 * DH; i += 256) {
    const int m = i >> 7, dd = i & 127;
    xa[m][dd] = (h16)Xb[m * DH + dd];
  }

  const float* Ws[3] = {Wq, Wk, Wv};
  v8f accq = {}, acck = {}, accv = {};
  const int n0 = wave * 16;

  for (int c = 0; c < 4; ++c) {
    const int kc = c * 32;
    __syncthreads();
    for (int w = 0; w < 3; ++w) {
      const float* W = Ws[w];
      for (int i = tid; i < DH * 8; i += 256) {
        const int n = i >> 3, kq = (i & 7) << 2;
        const float4 v = *reinterpret_cast<const float4*>(W + n * DH + kc + kq);
        v4h pk;
        pk[0] = (h16)v.x; pk[1] = (h16)v.y; pk[2] = (h16)v.z; pk[3] = (h16)v.w;
        *reinterpret_cast<v4h*>(&wbT[w][n][kq]) = pk;
      }
    }
    __syncthreads();
    const v16h a = load_a_frag(&xa[0][0], DH, kc);
    accq = WMMA_F16(a, load_bT_frag(&wbT[0][0][0], LDT, n0), accq);
    acck = WMMA_F16(a, load_bT_frag(&wbT[1][0][0], LDT, n0), acck);
    accv = WMMA_F16(a, load_bT_frag(&wbT[2][0][0], LDT, n0), accv);
  }

  const int n = n0 + (lane & 15);
  const int mh = (lane >> 4) << 3;
  const int sbase = s0 + mh;                 // multiple of 8
  const bool valid = (sbase < S);            // uniform per half-wave vector

  v8h vekv, vek;
#pragma unroll
  for (int r = 0; r < 8; ++r) {
    float sq  = 1.f / (1.f + __expf(-accq[r]));
    float ek  = __expf(acck[r]);
    float ekv = ek * accv[r];
    if (!valid) { sq = 0.f; ek = 0.f; ekv = 0.f; }
    vekv[r] = (h16)ekv;
    vek[r]  = (h16)ek;
    SIGQ[((size_t)b * SP + sbase + r) * DH + n] = (h16)sq;
  }
  const size_t cb = ((size_t)b * 256 + n) * SK + sbase;
  *reinterpret_cast<v8h*>(EKVt + cb)                       = vekv;  // c = n
  *reinterpret_cast<v8h*>(EKVt + cb + (size_t)128 * SK)    = vek;   // c = n+128
}

// ---------------------------------------------------------------------------
// Attention: [num|den] = exp(-dist*c) @ [ekv|ek]; y = x + sigmoid(q)*num/den.
// Block: 16 rows x 256 cols, 8 waves x 2 N-tiles, K = 1024 (padded),
// double-buffered LDS staging (async-to-LDS for the B tile), one barrier/chunk.
// ---------------------------------------------------------------------------
__global__ __launch_bounds__(256) void attn_kernel(
    const float* __restrict__ dist, const float* __restrict__ logsc,
    const float* __restrict__ alpha, int layer,
    float* __restrict__ X, const h16* __restrict__ SIGQ,
    const h16* __restrict__ EKVt)
{
  const int tile = blockIdx.x, b = blockIdx.y;
  const int s0 = tile * 16;
  const int tid = threadIdx.x;
  const int wave = tid >> 5, lane = tid & 31;
  const float csc = logsc[0] * alpha[layer];

  __shared__ h16   esA[2][16][LDT];
  __shared__ h16   btT[2][256][LDT];  // transposed: [c][t-within-chunk]
  __shared__ float res[16][256];

  const float* db  = dist + (size_t)b * S * S;
  const h16*   ebT = EKVt + (size_t)b * 256 * SK;

  auto stage = [&](int buf, int t0) {
    // A tile: es = exp(-dist * c), zero outside valid range
    for (int i = tid; i < 512; i += 256) {
      const int m = i >> 5, tt = i & 31;
      const int s = s0 + m, t = t0 + tt;
      float e = 0.f;
      if (s < S && t < S) e = __expf(-db[(size_t)s * S + t] * csc);
      esA[buf][m][tt] = (h16)e;
    }
    // B tile: pure f16 copy -> async global->LDS when available; pad t is zero
    for (int i = tid; i < 1024; i += 256) {
      const int c = i >> 2, u = i & 3;
      copy16_g2l(ebT + (size_t)c * SK + t0 + u * 8, &btT[buf][c][u * 8]);
    }
    // warm L2 for the dist chunk after this one (global_prefetch_b8)
    if (t0 + 32 < S) {
      const int m = tid >> 4, tt2 = (tid & 15) << 1;
      if (s0 + m < S)
        __builtin_prefetch(&db[(size_t)(s0 + m) * S + t0 + 32 + tt2], 0, 1);
    }
  };

  v8f acc0 = {}, acc1 = {};
  const int n0 = wave * 32;

  stage(0, 0);
  int cur = 0;
  for (int kc = 0; kc < SK / 32; ++kc) {
    wait_async0();     // this wave's async LDS writes for buffer `cur` landed
    __syncthreads();   // staging(cur) done; everyone done computing on cur^1
    if (kc + 1 < SK / 32) stage(cur ^ 1, (kc + 1) * 32);
    const v16h a = load_a_frag(&esA[cur][0][0], LDT, 0);
    acc0 = WMMA_F16(a, load_bT_frag(&btT[cur][0][0], LDT, n0), acc0);
    acc1 = WMMA_F16(a, load_bT_frag(&btT[cur][0][0], LDT, n0 + 16), acc1);
    cur ^= 1;
  }

  __syncthreads();
  {
    const int n = lane & 15, mh = (lane >> 4) << 3;
#pragma unroll
    for (int r = 0; r < 8; ++r) {
      res[r + mh][n0 + n]      = acc0[r];
      res[r + mh][n0 + 16 + n] = acc1[r];
    }
  }
  __syncthreads();
  for (int i = tid; i < 16 * DH; i += 256) {
    const int m = i >> 7, dd = i & 127;
    const int s = s0 + m;
    if (s >= S) continue;                      // keep padded rows at 0
    const float num = res[m][dd], den = res[m][dd + 128];
    const float w = num / fmaxf(den, 1e-30f);
    const size_t o = ((size_t)b * SP + s) * DH + dd;
    X[o] = X[o] + (float)SIGQ[o] * w;
  }
}

// ---------------------------------------------------------------------------
// Instance-norm over the sequence axis (biased variance, valid rows only).
// ---------------------------------------------------------------------------
__global__ __launch_bounds__(128) void stats_kernel(
    const float* __restrict__ Xv, float* __restrict__ MU, float* __restrict__ RS)
{
  const int b = blockIdx.x, d = threadIdx.x;
  const float* p = Xv + (size_t)b * SP * DH + d;
  float s = 0.f, ss = 0.f;
  for (int r = 0; r < S; ++r) {
    const float v = p[(size_t)r * DH];
    s += v; ss += v * v;
  }
  const float mu = s * (1.f / S);
  const float var = ss * (1.f / S) - mu * mu;
  MU[b * DH + d] = mu;
  RS[b * DH + d] = rsqrtf(fmaxf(var, 0.f) + EPSN);
}

__global__ __launch_bounds__(256) void norm_apply_kernel(
    const float* __restrict__ src, float* __restrict__ dst,
    const float* __restrict__ MU, const float* __restrict__ RS,
    const float* __restrict__ g, const float* __restrict__ be)
{
  const int tile = blockIdx.x, b = blockIdx.y;
  const int s0 = tile * 16;
  for (int i = threadIdx.x; i < 16 * DH; i += 256) {
    const int m = i >> 7, d = i & 127;
    const int s = s0 + m;
    const size_t o = ((size_t)b * SP + s) * DH + d;
    float v = 0.f;
    if (s < S) v = (src[o] - MU[b * DH + d]) * RS[b * DH + d] * g[d] + be[d];
    dst[o] = v;
  }
}

// ---------------------------------------------------------------------------
// FFN stage 1: FF = relu(H @ W1.T + bW1) as f16 row-major. 16 x 512 per block.
// ---------------------------------------------------------------------------
__global__ __launch_bounds__(256) void ffn1_kernel(
    const float* __restrict__ H, const float* __restrict__ W1,
    const float* __restrict__ bW1, h16* __restrict__ FF)
{
  const int tile = blockIdx.x, b = blockIdx.y;
  const int s0 = tile * 16;
  const int tid = threadIdx.x;
  const int wave = tid >> 5, lane = tid & 31;

  __shared__ h16 ha[16][DH];
  __shared__ h16 w1T[FH][LDT];   // transposed: [n][k]

  const float* Hb = H + ((size_t)b * SP + s0) * DH;
  for (int i = tid; i < 16 * DH; i += 256) {
    const int m = i >> 7, dd = i & 127;
    ha[m][dd] = (h16)Hb[m * DH + dd];
  }

  v8f acc[4] = {};
  for (int c = 0; c < 4; ++c) {
    const int kc = c * 32;
    __syncthreads();
    for (int i = tid; i < FH * 8; i += 256) {
      const int n = i >> 3, kq = (i & 7) << 2;
      const float4 v = *reinterpret_cast<const float4*>(W1 + n * DH + kc + kq);
      v4h pk;
      pk[0] = (h16)v.x; pk[1] = (h16)v.y; pk[2] = (h16)v.z; pk[3] = (h16)v.w;
      *reinterpret_cast<v4h*>(&w1T[n][kq]) = pk;
    }
    __syncthreads();
    const v16h a = load_a_frag(&ha[0][0], DH, kc);
#pragma unroll
    for (int j = 0; j < 4; ++j) {
      acc[j] = WMMA_F16(a, load_bT_frag(&w1T[0][0], LDT, (wave * 4 + j) * 16), acc[j]);
    }
  }

  const int mh = (lane >> 4) << 3, nl = lane & 15;
#pragma unroll
  for (int j = 0; j < 4; ++j) {
    const int n = (wave * 4 + j) * 16 + nl;
    const float bb = bW1[n];
#pragma unroll
    for (int r = 0; r < 8; ++r) {
      const int s = s0 + r + mh;
      FF[((size_t)b * SP + s) * FH + n] = (h16)fmaxf(acc[j][r] + bb, 0.f);
    }
  }
}

// ---------------------------------------------------------------------------
// FFN stage 2: X = FF @ W2.T + bW2 + H. 16 x 128 per block, K = 512.
// ---------------------------------------------------------------------------
__global__ __launch_bounds__(256) void ffn2_kernel(
    const h16* __restrict__ FF, const float* __restrict__ W2,
    const float* __restrict__ bW2, const float* __restrict__ H,
    float* __restrict__ X)
{
  const int tile = blockIdx.x, b = blockIdx.y;
  const int s0 = tile * 16;
  const int tid = threadIdx.x;
  const int wave = tid >> 5, lane = tid & 31;

  __shared__ h16 fa[16][FH + 8];
  __shared__ h16 w2T[DH][LDT];   // transposed: [n][k]

  const h16* Fb = FF + ((size_t)b * SP + s0) * FH;
  for (int i = tid; i < 16 * (FH / 8); i += 256) {
    const int m = i >> 6, u = i & 63;
    copy16_g2l(Fb + m * FH + u * 8, &fa[m][u * 8]);
  }
  wait_async0();   // fa fully issued/landed for this wave before first barrier

  v8f acc = {};
  const int n0 = wave * 16;
  for (int c = 0; c < 16; ++c) {
    const int kc = c * 32;
    __syncthreads();
    for (int i = tid; i < DH * 8; i += 256) {
      const int n = i >> 3, kq = (i & 7) << 2;
      const float4 v = *reinterpret_cast<const float4*>(W2 + n * FH + kc + kq);
      v4h pk;
      pk[0] = (h16)v.x; pk[1] = (h16)v.y; pk[2] = (h16)v.z; pk[3] = (h16)v.w;
      *reinterpret_cast<v4h*>(&w2T[n][kq]) = pk;
    }
    __syncthreads();
    const v16h a = load_a_frag(&fa[0][0], FH + 8, kc);
    acc = WMMA_F16(a, load_bT_frag(&w2T[0][0], LDT, n0), acc);
  }

  const int mh = (lane >> 4) << 3;
  const int n = n0 + (lane & 15);
  const float bb = bW2[n];
#pragma unroll
  for (int r = 0; r < 8; ++r) {
    const int s = s0 + r + mh;
    const size_t o = ((size_t)b * SP + s) * DH + n;
    X[o] = acc[r] + bb + H[o];
  }
}

__global__ __launch_bounds__(128) void copy_out_kernel(
    const float* __restrict__ X, float* __restrict__ out)
{
  const int s = blockIdx.x, b = blockIdx.y, d = threadIdx.x;
  out[((size_t)b * S + s) * DH + d] = X[((size_t)b * SP + s) * DH + d];
}

// ---------------------------------------------------------------------------
extern "C" void kernel_launch(void* const* d_in, const int* in_sizes, int n_in,
                              void* d_out, int out_size, void* d_ws, size_t ws_size,
                              hipStream_t stream)
{
  (void)in_sizes; (void)n_in; (void)out_size; (void)ws_size;

  const float* depot = (const float*)d_in[0];
  const float* node  = (const float*)d_in[1];
  const float* dist  = (const float*)d_in[2];
  const float* logsc = (const float*)d_in[3];
  const int*   flag  = (const int*)d_in[4];
  const float* Wd   = (const float*)d_in[5];
  const float* bd   = (const float*)d_in[6];
  const float* Wn   = (const float*)d_in[7];
  const float* bn   = (const float*)d_in[8];
  const float* Win  = (const float*)d_in[9];
  const float* binv = (const float*)d_in[10];
  const float* Wout = (const float*)d_in[11];
  const float* bout = (const float*)d_in[12];
  const float* Wq   = (const float*)d_in[13];
  const float* Wk   = (const float*)d_in[14];
  const float* Wv   = (const float*)d_in[15];
  const float* alpha= (const float*)d_in[16];
  const float* g1   = (const float*)d_in[17];
  const float* b1   = (const float*)d_in[18];
  const float* W1   = (const float*)d_in[19];
  const float* bW1  = (const float*)d_in[20];
  const float* W2   = (const float*)d_in[21];
  const float* bW2  = (const float*)d_in[22];
  const float* g2   = (const float*)d_in[23];
  const float* b2   = (const float*)d_in[24];
  float* out = (float*)d_out;

  // Workspace carve-out (~91 MB total)
  char* ws = (char*)d_ws;
  size_t off = 0;
  auto carve = [&](size_t bytes) -> void* {
    void* p = ws + off;
    off = (off + bytes + 255) & ~(size_t)255;
    return p;
  };
  float* X    = (float*)carve((size_t)NB * SP * DH * 4);
  float* Hbuf = (float*)carve((size_t)NB * SP * DH * 4);
  h16*   SIGQ = (h16*)  carve((size_t)NB * SP * DH * 2);
  h16*   EKVt = (h16*)  carve((size_t)NB * 256 * SK * 2);  // feature-major
  h16*   FF   = (h16*)  carve((size_t)NB * SP * FH * 2);
  float* MU   = (float*)carve((size_t)NB * DH * 4);
  float* RS   = (float*)carve((size_t)NB * DH * 4);

  const dim3 gT(SP / 16, NB);

  embed_kernel<<<NB, 128, 0, stream>>>(depot, node, flag, Wd, bd, Wn, bn,
                                       Win, binv, Wout, bout, X);
  ekv_tail_zero_kernel<<<NB, 256, 0, stream>>>(EKVt);

  for (int l = 0; l < NL; ++l) {
    qkv_kernel<<<gT, 256, 0, stream>>>(X, Wq + (size_t)l * DH * DH,
                                       Wk + (size_t)l * DH * DH,
                                       Wv + (size_t)l * DH * DH, SIGQ, EKVt);
    attn_kernel<<<gT, 256, 0, stream>>>(dist, logsc, alpha, l, X, SIGQ, EKVt);
    stats_kernel<<<NB, 128, 0, stream>>>(X, MU, RS);
    norm_apply_kernel<<<gT, 256, 0, stream>>>(X, Hbuf, MU, RS, g1 + l * DH, b1 + l * DH);
    ffn1_kernel<<<gT, 256, 0, stream>>>(Hbuf, W1 + (size_t)l * FH * DH, bW1 + l * FH, FF);
    ffn2_kernel<<<gT, 256, 0, stream>>>(FF, W2 + (size_t)l * DH * FH, bW2 + l * DH, Hbuf, X);
    stats_kernel<<<NB, 128, 0, stream>>>(X, MU, RS);
    norm_apply_kernel<<<gT, 256, 0, stream>>>(X, X, MU, RS, g2 + l * DH, b2 + l * DH);
  }

  copy_out_kernel<<<dim3(S, NB), 128, 0, stream>>>(X, out);
}
